// MultiheadAttention_40707700031881
// MI455X (gfx1250) — matrix operand order repaired and checked by reading, not compiled, not run on hardware
//
#include <hip/hip_runtime.h>
#include <hip/hip_bf16.h>
#include <math.h>

typedef __attribute__((ext_vector_type(16))) __bf16 v16bf;
typedef __attribute__((ext_vector_type(8)))  float  v8f;
typedef __attribute__((ext_vector_type(2)))  __bf16 bf16x2;
typedef __attribute__((ext_vector_type(4)))  unsigned int u32x4;
typedef __attribute__((ext_vector_type(8)))  int i32x8;
typedef __attribute__((ext_vector_type(4)))  int i32x4;

// ---------------------------------------------------------------------------
// Tensor Data Mover: issue a 2-D tile load (global -> LDS).
// D# packing per CDNA5 ISA:
//   group0: [1:0]=count=1, [63:32]=lds_addr, [120:64]=global_addr, [127:126]=type=2
//   group1: [17:16]=data_size(1 => 2 bytes), [79:48]=tensor_dim0, [111:80]=tensor_dim1,
//           [127:112]=tile_dim0, [143:128]=tile_dim1, [207:160]=tensor_dim0_stride
// Dims/strides in data_size units. Groups 2/3 zero (2-D tensor).
// This toolchain exposes the 6-arg builtin: (u32x4, i32x8, i32x4, i32x4, i32x8, cpol).
// ---------------------------------------------------------------------------
__device__ __forceinline__ unsigned lds_addr_of(const void* p) {
    return (unsigned)(uintptr_t)p;   // low 32 bits of shared-aperture address = LDS offset
}

__device__ __forceinline__ void tdm_load_2d_bf16(unsigned lds_addr, const void* gptr,
                                                 unsigned tile_x, unsigned tile_y,
                                                 unsigned tensor_x, unsigned tensor_y,
                                                 unsigned stride_x) {
    const unsigned long long ga = (unsigned long long)gptr;
    u32x4 g0;
    g0[0] = 1u;                                             // count=1 (valid), user mode
    g0[1] = lds_addr;                                       // LDS byte address
    g0[2] = (unsigned)(ga & 0xFFFFFFFFu);                   // global_addr[31:0]
    g0[3] = (unsigned)((ga >> 32) & 0x01FFFFFFu) | (2u << 30); // global_addr[56:32] | type=2
    i32x8 g1;
    g1[0] = (int)(1u << 16);                                // data_size=1 (2 bytes), wg_mask=0
    g1[1] = (int)((tensor_x & 0xFFFFu) << 16);              // tensor_dim0[15:0]
    g1[2] = (int)(((tensor_x >> 16) & 0xFFFFu) | ((tensor_y & 0xFFFFu) << 16));
    g1[3] = (int)(((tensor_y >> 16) & 0xFFFFu) | ((tile_x & 0xFFFFu) << 16));
    g1[4] = (int)(tile_y & 0xFFFFu);                        // tile_dim1 (tile_dim2=0)
    g1[5] = (int)stride_x;                                  // tensor_dim0_stride[31:0]
    g1[6] = 0;
    g1[7] = 0;
    i32x4 gz4 = {0, 0, 0, 0};
    i32x8 gz8 = {0, 0, 0, 0, 0, 0, 0, 0};
    __builtin_amdgcn_tensor_load_to_lds(g0, g1, gz4, gz4, gz8, 0);
}

// ---------------------------------------------------------------------------
// fp32 -> bf16 elementwise convert
// ---------------------------------------------------------------------------
__global__ void cvt_f32_to_bf16_kernel(const float* __restrict__ src,
                                       __bf16* __restrict__ dst, int n) {
    int i = blockIdx.x * blockDim.x + threadIdx.x;
    if (i < n) dst[i] = (__bf16)src[i];
}

// ---------------------------------------------------------------------------
// TDM-staged bf16 GEMM: C[M,N] = A[M,K] * B[K,N] + bias[N]
// 256 threads = 8 waves as 2(M) x 4(N); block tile 64M x 64N, K-step 32.
// Wave 0 DMA-stages A(64x32) and B(32x64) tiles into LDS via the Tensor Data
// Mover, waits on TENSORcnt, then all waves consume fragments from LDS.
// ---------------------------------------------------------------------------
template <bool OUT_BF16>
__global__ __launch_bounds__(256)
void wmma_gemm_tdm_kernel(const __bf16* __restrict__ A, const __bf16* __restrict__ B,
                          const float* __restrict__ bias,
                          float* __restrict__ Cf, __bf16* __restrict__ Cb,
                          int M, int N, int K) {
    __shared__ __bf16 Ash[64][32];   // 4 KB
    __shared__ __bf16 Bsh[32][64];   // 4 KB
    const int lane = threadIdx.x & 31;
    const int wave = threadIdx.x >> 5;
    const int wm = wave >> 2;        // 0..1
    const int wn = wave & 3;         // 0..3
    const int l15 = lane & 15;
    const int half = lane >> 4;

    const int mBlock = blockIdx.y * 64;
    const int nBlock = blockIdx.x * 64;
    const int nl = wn * 16 + l15;    // column within block tile
    const int n  = nBlock + nl;      // global column

    v8f acc0 = {}, acc1 = {};
    for (int k0 = 0; k0 < K; k0 += 32) {
        if (threadIdx.x < 32) {
            tdm_load_2d_bf16(lds_addr_of(&Ash[0][0]), A + (size_t)mBlock * K + k0,
                             /*tile*/ 32, 64, /*tensor*/ (unsigned)K, (unsigned)M,
                             /*stride*/ (unsigned)K);
            tdm_load_2d_bf16(lds_addr_of(&Bsh[0][0]), B + (size_t)k0 * N + nBlock,
                             /*tile*/ 64, 32, /*tensor*/ (unsigned)N, (unsigned)K,
                             /*stride*/ (unsigned)N);
            __builtin_amdgcn_s_wait_tensorcnt(0);
        }
        __syncthreads();

        v16bf bfrag;
#pragma unroll
        for (int j = 0; j < 8; ++j) {
            bfrag[2 * j]     = Bsh[16 * half + 2 * j][nl];
            bfrag[2 * j + 1] = Bsh[16 * half + 2 * j + 1][nl];
        }
#pragma unroll
        for (int t = 0; t < 2; ++t) {
            v16bf afrag;
            const __bf16* Ap = &Ash[wm * 32 + t * 16 + l15][16 * half];
#pragma unroll
            for (int j = 0; j < 4; ++j) {
                bf16x2 p = *(const bf16x2*)(Ap + 2 * j);
                afrag[2 * j] = p.x; afrag[2 * j + 1] = p.y;
            }
#pragma unroll
            for (int j = 4; j < 8; ++j) {
                bf16x2 p = *(const bf16x2*)(Ap + 2 * j);
                afrag[2 * j] = p.x; afrag[2 * j + 1] = p.y;
            }
            if (t == 0)
                acc0 = __builtin_amdgcn_wmma_f32_16x16x32_bf16(false, afrag, false, bfrag,
                                                               (short)0, acc0, false, false);
            else
                acc1 = __builtin_amdgcn_wmma_f32_16x16x32_bf16(false, afrag, false, bfrag,
                                                               (short)0, acc1, false, false);
        }
        __syncthreads();
    }

    const float bv = bias[n];
#pragma unroll
    for (int i = 0; i < 8; ++i) {
        const int m0 = mBlock + wm * 32 + i + 8 * half;
        const float v0 = acc0[i] + bv;
        const float v1 = acc1[i] + bv;
        if (OUT_BF16) {
            Cb[(size_t)m0 * N + n]        = (__bf16)v0;
            Cb[(size_t)(m0 + 16) * N + n] = (__bf16)v1;
        } else {
            Cf[(size_t)m0 * N + n]        = v0;
            Cf[(size_t)(m0 + 16) * N + n] = v1;
        }
    }
}

// ---------------------------------------------------------------------------
// Attention logits: per (b,h): Q (S x 64) * K^T / 8 -> fp32.
// Q/K rows are contiguous -> direct global_load_b128 into fragments (proven
// good codegen in round 1); 2 WMMAs per 16x16 tile.
// ---------------------------------------------------------------------------
__global__ void attn_logits_wmma_kernel(const __bf16* __restrict__ qkv,
                                        float* __restrict__ attn) {
    const int lane = threadIdx.x & 31;
    const int wave = threadIdx.x >> 5;
    const int l15  = lane & 15;
    const int half = lane >> 4;

    const unsigned tile  = blockIdx.x * 4 + wave;   // 0 .. 262143
    const int ktile = tile & 63;
    const int qtile = (tile >> 6) & 63;
    const int bh    = tile >> 12;                   // b*16 + h
    const int h = bh & 15;
    const int b = bh >> 4;

    const __bf16* base = qkv + (size_t)b * 1024 * 3072 + h * 192;
    const __bf16* Qrow = base + (size_t)(qtile * 16 + l15) * 3072;        // Q
    const __bf16* Krow = base + (size_t)(ktile * 16 + l15) * 3072 + 64;   // K

    v8f acc = {};
#pragma unroll
    for (int k0 = 0; k0 < 64; k0 += 32) {
        v16bf a, bf;
        const __bf16* Ap = Qrow + k0 + 16 * half;
#pragma unroll
        for (int j = 0; j < 4; ++j) {
            bf16x2 p = *(const bf16x2*)(Ap + 2 * j);
            a[2 * j] = p.x; a[2 * j + 1] = p.y;
        }
#pragma unroll
        for (int j = 4; j < 8; ++j) {
            bf16x2 p = *(const bf16x2*)(Ap + 2 * j);
            a[2 * j] = p.x; a[2 * j + 1] = p.y;
        }
        const __bf16* Bp = Krow + k0 + 16 * half;
#pragma unroll
        for (int j = 0; j < 8; ++j) {
            bf16x2 p = *(const bf16x2*)(Bp + 2 * j);
            bf[2 * j] = p.x; bf[2 * j + 1] = p.y;
        }
        acc = __builtin_amdgcn_wmma_f32_16x16x32_bf16(false, a, false, bf,
                                                      (short)0, acc, false, false);
    }

    float* out = attn + ((size_t)bh * 1024 + qtile * 16) * 1024 + ktile * 16 + l15;
#pragma unroll
    for (int i = 0; i < 8; ++i)
        out[(size_t)(i + 8 * half) * 1024] = acc[i] * 0.125f;  // 1/sqrt(64)
}

// ---------------------------------------------------------------------------
// In-place row softmax over 1024 columns. One 256-thread block per row.
// ---------------------------------------------------------------------------
__global__ void softmax1024_kernel(float* __restrict__ attn) {
    float* p = attn + (size_t)blockIdx.x * 1024;
    const int t = threadIdx.x;
    const int lane = t & 31;
    const int w = t >> 5;
    __shared__ float smax[8];
    __shared__ float ssum[8];

    float v[4];
    float mx = -INFINITY;
#pragma unroll
    for (int i = 0; i < 4; ++i) {
        v[i] = p[t + 256 * i];
        mx = fmaxf(mx, v[i]);
    }
#pragma unroll
    for (int off = 16; off > 0; off >>= 1)
        mx = fmaxf(mx, __shfl_xor(mx, off, 32));
    if (lane == 0) smax[w] = mx;
    __syncthreads();
    float m = smax[0];
#pragma unroll
    for (int i = 1; i < 8; ++i) m = fmaxf(m, smax[i]);

    float s = 0.0f;
#pragma unroll
    for (int i = 0; i < 4; ++i) {
        v[i] = __expf(v[i] - m);
        s += v[i];
    }
#pragma unroll
    for (int off = 16; off > 0; off >>= 1)
        s += __shfl_xor(s, off, 32);
    if (lane == 0) ssum[w] = s;
    __syncthreads();
    float tot = 0.0f;
#pragma unroll
    for (int i = 0; i < 8; ++i) tot += ssum[i];
    const float inv = 1.0f / tot;
#pragma unroll
    for (int i = 0; i < 4; ++i) p[t + 256 * i] = v[i] * inv;
}

// ---------------------------------------------------------------------------
// values = attention * V -> bf16 [B,S,D].
// 128 threads = 4 waves; block owns (bh, qtile), waves own the 4 hd-tiles.
// V tile (32 seq rows x 64 features, global stride 3072) is DMA-staged into
// LDS by the TDM; attention rows stay on the direct b128+cvt path.
// ---------------------------------------------------------------------------
__global__ __launch_bounds__(128)
void attn_v_tdm_kernel(const float* __restrict__ attn,
                       const __bf16* __restrict__ qkv,
                       __bf16* __restrict__ valb) {
    __shared__ __bf16 Vsh[32][64];   // 4 KB
    const int lane = threadIdx.x & 31;
    const int wave = threadIdx.x >> 5;   // = hd tile index, 0..3
    const int l15  = lane & 15;
    const int half = lane >> 4;

    const int bq    = blockIdx.x;        // bh*64 + qtile
    const int qtile = bq & 63;
    const int bh    = bq >> 6;
    const int h = bh & 15;
    const int b = bh >> 4;

    const int qrow = qtile * 16 + l15;
    const int nl   = wave * 16 + l15;
    const float*  Arow  = attn + ((size_t)bh * 1024 + qrow) * 1024;
    const __bf16* Vbase = qkv + (size_t)b * 1024 * 3072 + h * 192 + 128;

    v8f acc = {};
    for (int k0 = 0; k0 < 1024; k0 += 32) {
        if (threadIdx.x < 32) {
            tdm_load_2d_bf16(lds_addr_of(&Vsh[0][0]), Vbase + (size_t)k0 * 3072,
                             /*tile*/ 64, 32, /*tensor*/ 64, 1024, /*stride*/ 3072);
            __builtin_amdgcn_s_wait_tensorcnt(0);
        }
        __syncthreads();

        v16bf a, bf;
        const float* Ap = Arow + k0 + 16 * half;
#pragma unroll
        for (int j = 0; j < 4; ++j) {
            a[2 * j]     = (__bf16)Ap[2 * j];
            a[2 * j + 1] = (__bf16)Ap[2 * j + 1];
        }
#pragma unroll
        for (int j = 4; j < 8; ++j) {
            a[2 * j]     = (__bf16)Ap[2 * j];
            a[2 * j + 1] = (__bf16)Ap[2 * j + 1];
        }
#pragma unroll
        for (int j = 0; j < 8; ++j) {
            bf[2 * j]     = Vsh[16 * half + 2 * j][nl];
            bf[2 * j + 1] = Vsh[16 * half + 2 * j + 1][nl];
        }
        acc = __builtin_amdgcn_wmma_f32_16x16x32_bf16(false, a, false, bf,
                                                      (short)0, acc, false, false);
        __syncthreads();
    }

    __bf16* out = valb + ((size_t)b * 1024 + qtile * 16) * 1024 + h * 64 + wave * 16 + l15;
#pragma unroll
    for (int i = 0; i < 8; ++i)
        out[(size_t)(i + 8 * half) * 1024] = (__bf16)acc[i];
}

// ---------------------------------------------------------------------------
extern "C" void kernel_launch(void* const* d_in, const int* in_sizes, int n_in,
                              void* d_out, int out_size, void* d_ws, size_t ws_size,
                              hipStream_t stream) {
    const float* x     = (const float*)d_in[0];   // [4,1024,1024]
    const float* W_qkv = (const float*)d_in[1];   // [1024,3072]
    const float* b_qkv = (const float*)d_in[2];   // [3072]
    const float* W_o   = (const float*)d_in[3];   // [1024,1024]
    const float* b_o   = (const float*)d_in[4];   // [1024]

    float* o_out    = (float*)d_out;                         // 4M floats
    float* attn_out = o_out + (size_t)4 * 1024 * 1024;       // 64M floats

    char* ws = (char*)d_ws;
    __bf16* xb    = (__bf16*)(ws);                             // 8 MB
    __bf16* wqkvb = (__bf16*)(ws + ((size_t)8  << 20));        // 6 MB
    __bf16* wob   = (__bf16*)(ws + ((size_t)14 << 20));        // 2 MB
    __bf16* qkvb  = (__bf16*)(ws + ((size_t)16 << 20));        // 24 MB
    __bf16* valb  = (__bf16*)(ws + ((size_t)40 << 20));        // 8 MB  (total 48 MB)

    // 1. fp32 -> bf16 converts
    cvt_f32_to_bf16_kernel<<<(4194304 + 255) / 256, 256, 0, stream>>>(x, xb, 4194304);
    cvt_f32_to_bf16_kernel<<<(3145728 + 255) / 256, 256, 0, stream>>>(W_qkv, wqkvb, 3145728);
    cvt_f32_to_bf16_kernel<<<(1048576 + 255) / 256, 256, 0, stream>>>(W_o, wob, 1048576);

    // 2. QKV GEMM: [4096,1024] x [1024,3072] + b_qkv -> bf16 qkv (TDM-staged)
    wmma_gemm_tdm_kernel<true><<<dim3(3072 / 64, 4096 / 64), 256, 0, stream>>>(
        xb, wqkvb, b_qkv, nullptr, qkvb, 4096, 3072, 1024);

    // 3. Logits Q.K^T / 8 straight into the attention output slice
    attn_logits_wmma_kernel<<<65536, 128, 0, stream>>>(qkvb, attn_out);

    // 4. Row softmax in place (65536 rows of 1024)
    softmax1024_kernel<<<65536, 256, 0, stream>>>(attn_out);

    // 5. values = attn @ V -> bf16 [B,S,D] (V tiles TDM-staged)
    attn_v_tdm_kernel<<<4096, 128, 0, stream>>>(attn_out, qkvb, valb);

    // 6. Output projection: [4096,1024] x [1024,1024] + b_o -> o (fp32, TDM-staged)
    wmma_gemm_tdm_kernel<false><<<dim3(1024 / 64, 4096 / 64), 256, 0, stream>>>(
        valb, wob, b_o, o_out, nullptr, 4096, 1024, 1024);
}
